// FastMQAttention_59390807769258
// MI455X (gfx1250) — matrix operand order repaired and checked by reading, not compiled, not run on hardware
//
#include <hip/hip_runtime.h>
#include <hip/hip_bf16.h>

typedef __attribute__((ext_vector_type(16))) _Float16 v16h;
typedef __attribute__((ext_vector_type(8)))  _Float16 h8;
typedef __attribute__((ext_vector_type(4)))  _Float16 h4;
typedef __attribute__((ext_vector_type(8)))  float    v8f;
typedef __attribute__((ext_vector_type(4)))  float    vf4;

#define NUM_HEADS 32
#define HEAD_DIM  128
#define SEQ       2048
#define ATTN_SCALE 0.08838834764831845f  /* 1/sqrt(128) */

#define BQ 128           /* query rows per block (8 waves x 16) */
#define BK 32            /* key rows per iteration               */
#define KSTRIDE 136      /* halves per K_lds row   (272 B, 16-aligned) */
#define VSTRIDE 40       /* halves per VT_lds row  (80 B,  16-aligned) */
#define SSTRIDE 40       /* floats per S_lds row   (160 B, 16-aligned) */
#define PSTRIDE 40       /* halves per P_lds row   (80 B,  16-aligned) */

__global__ __launch_bounds__(256)
void mqa_flash_kernel(const float* __restrict__ Q, const float* __restrict__ K,
                      const float* __restrict__ V, float* __restrict__ O)
{
    __shared__ _Float16 K_lds[BK * KSTRIDE];          /* 8704 B  : 32 keys x 128 d (f16)      */
    __shared__ _Float16 VT_lds[HEAD_DIM * VSTRIDE];   /* 10240 B : 128 d x 32 keys (f16, T)   */
    __shared__ float    S_lds[8 * 16 * SSTRIDE];      /* 20480 B : per-wave raw scores (f32)  */
    __shared__ _Float16 P_lds[8 * 16 * PSTRIDE];      /* 10240 B : per-wave probs (f16)       */
    __shared__ float    F_lds[8 * 16];                /* 512 B   : per-wave row factors       */

    const int tid  = threadIdx.x;       /* 0..255 */
    const int w    = tid >> 5;          /* wave id 0..7 */
    const int lane = tid & 31;
    const int r    = lane & 15;         /* row/col within 16 */
    const int hh   = lane >> 4;         /* lane-half select  */

    const int bh    = blockIdx.x >> 4;  /* 0..63 : b*32 + h  */
    const int chunk = blockIdx.x & 15;  /* 128-row Q chunk   */
    const int b     = bh >> 5;

    const float* Qb = Q + ((size_t)bh * SEQ + (size_t)chunk * BQ) * HEAD_DIM;
    const float* Kb = K + (size_t)b * SEQ * HEAD_DIM;
    const float* Vb = V + (size_t)b * SEQ * HEAD_DIM;
    float*       Ob = O + ((size_t)bh * SEQ + (size_t)chunk * BQ) * HEAD_DIM;

    /* ---- Q tile resident in registers as 4 A-fragments (f16, A-layout) ---- */
    v16h aQ[4];
    {
        const float* qrow = Qb + (size_t)(w * 16 + r) * HEAD_DIM;
        #pragma unroll
        for (int f = 0; f < 4; ++f) {
            const int d0 = f * 32 + hh * 8;               /* K = hh*8 + i            */
            vf4 x0 = *(const vf4*)(qrow + d0);
            vf4 x1 = *(const vf4*)(qrow + d0 + 4);
            vf4 x2 = *(const vf4*)(qrow + d0 + 16);       /* K = 16 + hh*8 + i       */
            vf4 x3 = *(const vf4*)(qrow + d0 + 20);
            v16h a;
            a[0]=(_Float16)x0[0];  a[1]=(_Float16)x0[1];  a[2]=(_Float16)x0[2];  a[3]=(_Float16)x0[3];
            a[4]=(_Float16)x1[0];  a[5]=(_Float16)x1[1];  a[6]=(_Float16)x1[2];  a[7]=(_Float16)x1[3];
            a[8]=(_Float16)x2[0];  a[9]=(_Float16)x2[1];  a[10]=(_Float16)x2[2]; a[11]=(_Float16)x2[3];
            a[12]=(_Float16)x3[0]; a[13]=(_Float16)x3[1]; a[14]=(_Float16)x3[2]; a[15]=(_Float16)x3[3];
            aQ[f] = a;
        }
    }

    /* ---- flash-attention running state ---- */
    v8f acc[8];
    #pragma unroll
    for (int t = 0; t < 8; ++t)
        #pragma unroll
        for (int j = 0; j < 8; ++j) acc[t][j] = 0.0f;
    float m_run = -INFINITY;   /* valid in lanes 0..15 */
    float l_run = 0.0f;

    float*     Sw = &S_lds[w * 16 * SSTRIDE];
    _Float16*  Pw = &P_lds[w * 16 * PSTRIDE];
    float*     Fw = &F_lds[w * 16];

    #pragma clang loop unroll(disable)
    for (int it = 0; it < SEQ / BK; ++it) {
        const int kb = it * BK;

        __syncthreads();   /* previous iteration done reading K/V LDS */

        /* ---- cooperative stage: K row-major f16, V transposed f16 ---- */
        #pragma unroll
        for (int i = 0; i < 4; ++i) {
            const int e   = (i * 256 + tid) * 4;   /* element in 32x128 tile */
            const int row = e >> 7;
            const int col = e & 127;
            vf4 kv = *(const vf4*)(Kb + (size_t)(kb + row) * HEAD_DIM + col);
            h4 kh = { (_Float16)kv[0], (_Float16)kv[1], (_Float16)kv[2], (_Float16)kv[3] };
            *(h4*)&K_lds[row * KSTRIDE + col] = kh;
            vf4 vv = *(const vf4*)(Vb + (size_t)(kb + row) * HEAD_DIM + col);
            VT_lds[(col + 0) * VSTRIDE + row] = (_Float16)vv[0];
            VT_lds[(col + 1) * VSTRIDE + row] = (_Float16)vv[1];
            VT_lds[(col + 2) * VSTRIDE + row] = (_Float16)vv[2];
            VT_lds[(col + 3) * VSTRIDE + row] = (_Float16)vv[3];
        }
        __syncthreads();

        /* ---- scores: S(16x32) = Qtile(16x128) @ Kblk^T(128x32), 8 WMMA ---- */
        v8f c0, c1;
        #pragma unroll
        for (int j = 0; j < 8; ++j) { c0[j] = 0.0f; c1[j] = 0.0f; }
        #pragma unroll
        for (int f = 0; f < 4; ++f) {
            /* B-fragment: lane = column(key); K(d) interleaved by lane half (hh*8, 16+hh*8) */
            const int d0 = f * 32 + hh * 8;
            v16h bk0, bk1;
            ((h8*)&bk0)[0] = *(const h8*)&K_lds[r        * KSTRIDE + d0];
            ((h8*)&bk0)[1] = *(const h8*)&K_lds[r        * KSTRIDE + d0 + 16];
            ((h8*)&bk1)[0] = *(const h8*)&K_lds[(16 + r) * KSTRIDE + d0];
            ((h8*)&bk1)[1] = *(const h8*)&K_lds[(16 + r) * KSTRIDE + d0 + 16];
            c0 = __builtin_amdgcn_wmma_f32_16x16x32_f16(false, aQ[f], false, bk0,
                                                        (short)0, c0, false, false);
            c1 = __builtin_amdgcn_wmma_f32_16x16x32_f16(false, aQ[f], false, bk1,
                                                        (short)0, c1, false, false);
        }

        /* ---- spill scores to per-wave LDS (C-layout -> row-major) ---- */
        #pragma unroll
        for (int j = 0; j < 8; ++j) {
            Sw[(j + 8 * hh) * SSTRIDE + r]      = c0[j];
            Sw[(j + 8 * hh) * SSTRIDE + 16 + r] = c1[j];
        }
        asm volatile("s_wait_dscnt 0" ::: "memory");

        /* ---- online softmax: lanes 0..15 each own one row ---- */
        if (lane < 16) {
            float s[32];
            float mt = -INFINITY;
            #pragma unroll
            for (int i = 0; i < 8; ++i) {
                vf4 v = *(const vf4*)&Sw[r * SSTRIDE + i * 4];
                s[4*i+0] = v[0] * ATTN_SCALE;
                s[4*i+1] = v[1] * ATTN_SCALE;
                s[4*i+2] = v[2] * ATTN_SCALE;
                s[4*i+3] = v[3] * ATTN_SCALE;
                mt = fmaxf(mt, fmaxf(fmaxf(s[4*i+0], s[4*i+1]), fmaxf(s[4*i+2], s[4*i+3])));
            }
            const float m_new = fmaxf(m_run, mt);
            const float corr  = __expf(m_run - m_new);
            m_run = m_new;
            float sum = 0.0f;
            #pragma unroll
            for (int i = 0; i < 8; ++i) {
                float p0 = __expf(s[4*i+0] - m_new);
                float p1 = __expf(s[4*i+1] - m_new);
                float p2 = __expf(s[4*i+2] - m_new);
                float p3 = __expf(s[4*i+3] - m_new);
                sum += (p0 + p1) + (p2 + p3);
                h4 ph = { (_Float16)p0, (_Float16)p1, (_Float16)p2, (_Float16)p3 };
                *(h4*)&Pw[r * PSTRIDE + i * 4] = ph;
            }
            l_run = l_run * corr + sum;
            Fw[r] = corr;
        }
        asm volatile("s_wait_dscnt 0" ::: "memory");

        /* ---- broadcast per-row rescale factors (C-layout rows) ---- */
        vf4 f0 = *(const vf4*)&Fw[8 * hh];
        vf4 f1 = *(const vf4*)&Fw[8 * hh + 4];
        float fac[8] = { f0[0], f0[1], f0[2], f0[3], f1[0], f1[1], f1[2], f1[3] };

        /* ---- P as A-fragment (interleaved A-layout) ---- */
        v16h aP;
        ((h8*)&aP)[0] = *(const h8*)&Pw[r * PSTRIDE + hh * 8];
        ((h8*)&aP)[1] = *(const h8*)&Pw[r * PSTRIDE + 16 + hh * 8];

        /* ---- O = diag(corr)*O + P @ Vblk : 8 WMMA over d-tiles ---- */
        #pragma unroll
        for (int t = 0; t < 8; ++t) {
            #pragma unroll
            for (int j = 0; j < 8; ++j) acc[t][j] *= fac[j];
            v16h bv;   /* lane = column d; K(key) interleaved by lane half (hh*8, 16+hh*8) */
            ((h8*)&bv)[0] = *(const h8*)&VT_lds[(t * 16 + r) * VSTRIDE + hh * 8];
            ((h8*)&bv)[1] = *(const h8*)&VT_lds[(t * 16 + r) * VSTRIDE + 16 + hh * 8];
            acc[t] = __builtin_amdgcn_wmma_f32_16x16x32_f16(false, aP, false, bv,
                                                            (short)0, acc[t], false, false);
        }
    }

    /* ---- final normalization by 1/l and store ---- */
    if (lane < 16) Fw[r] = 1.0f / l_run;
    asm volatile("s_wait_dscnt 0" ::: "memory");
    vf4 g0 = *(const vf4*)&Fw[8 * hh];
    vf4 g1 = *(const vf4*)&Fw[8 * hh + 4];
    float inv[8] = { g0[0], g0[1], g0[2], g0[3], g1[0], g1[1], g1[2], g1[3] };

    #pragma unroll
    for (int t = 0; t < 8; ++t)
        #pragma unroll
        for (int j = 0; j < 8; ++j)
            Ob[(size_t)(w * 16 + j + 8 * hh) * HEAD_DIM + t * 16 + r] = acc[t][j] * inv[j];
}

extern "C" void kernel_launch(void* const* d_in, const int* in_sizes, int n_in,
                              void* d_out, int out_size, void* d_ws, size_t ws_size,
                              hipStream_t stream) {
    const float* Q = (const float*)d_in[0];
    const float* K = (const float*)d_in[1];
    const float* V = (const float*)d_in[2];
    float*       O = (float*)d_out;

    /* grid: (B*H) * (S/BQ) = 64 * 16 = 1024 blocks, 256 threads (8 wave32) */
    dim3 grid(1024), block(256);
    mqa_flash_kernel<<<grid, block, 0, stream>>>(Q, K, V, O);
}
